// MDTA_91156385890963
// MI455X (gfx1250) — matrix operand order repaired and checked by reading, not compiled, not run on hardware
//
#include <hip/hip_runtime.h>

// ---------------- problem constants ----------------
#define kB     8
#define kC     192
#define kH     128
#define kW     128
#define kHW    (kH * kW)        // 16384
#define kT     kHW              // tokens per batch (t = w*H + h)
#define kNTok  (kB * kT)        // 131072
#define kHeads 4
#define kHS    48               // head dim
#define kKSplit 64              // K-split for the K^T Q reduction GEMM
#define kEPS   1e-5f
#define kHalfK 96               // weight staging: half of K per LDS stage

// ---------------- WMMA types ----------------
typedef __attribute__((ext_vector_type(16))) __bf16 v16bf;
typedef __attribute__((ext_vector_type(8)))  float  v8f;
typedef __attribute__((ext_vector_type(4)))  unsigned int u32x4;
typedef __attribute__((ext_vector_type(8)))  int i32x8;
typedef __attribute__((ext_vector_type(4)))  int i32x4;

union FragU { v16bf v; uint4 q[2]; };

static __device__ __forceinline__ __bf16 f2bf(float f) {
  unsigned int u = __builtin_bit_cast(unsigned int, f);
  u += 0x7fffu + ((u >> 16) & 1u);               // round-to-nearest-even
  unsigned short h = (unsigned short)(u >> 16);
  return __builtin_bit_cast(__bf16, h);
}

// A fragment: 16x32 bf16, row-major source, lda in elements.
// Lanes 0-15 row M=lane, elems0-7 = K0..7, elems8-15 = K16..23
// Lanes16-31 row M=lane-16, elems0-7 = K8..15, elems8-15 = K24..31
static __device__ __forceinline__ v16bf load_a_frag(const __bf16* __restrict__ base,
                                                    int row0, int lda, int k0, int lane) {
  const __bf16* p = base + (size_t)(row0 + (lane & 15)) * (size_t)lda
                         + (size_t)(k0 + ((lane & 16) ? 8 : 0));
  FragU u;
  u.q[0] = *reinterpret_cast<const uint4*>(p);
  u.q[1] = *reinterpret_cast<const uint4*>(p + 16);
  return u.v;
}

// B fragment from row-major B^T (N x K) storage, ldb in elements (global memory).
// Lanes 0-15 col N=lane, k = k0..k0+7, k0+16..k0+23 ; lanes16-31 col N=lane-16, k = k0+8..k0+15, k0+24..k0+31
static __device__ __forceinline__ v16bf load_b_frag(const __bf16* __restrict__ bt,
                                                    int n0, int ldb, int k0, int lane) {
  const __bf16* p = bt + (size_t)(n0 + (lane & 15)) * (size_t)ldb
                       + (size_t)(k0 + ((lane & 16) ? 8 : 0));
  FragU u;
  u.q[0] = *reinterpret_cast<const uint4*>(p);
  u.q[1] = *reinterpret_cast<const uint4*>(p + 16);
  return u.v;
}

// B fragment from LDS-staged weights, row-major (N x kHalfK)
static __device__ __forceinline__ v16bf load_b_frag_lds(const __bf16* sm,
                                                        int n0, int k0, int lane) {
  const __bf16* p = sm + (n0 + (lane & 15)) * kHalfK + k0 + ((lane & 16) ? 8 : 0);
  FragU u;
  u.q[0] = *reinterpret_cast<const uint4*>(p);
  u.q[1] = *reinterpret_cast<const uint4*>(p + 16);
  return u.v;
}

static __device__ __forceinline__ v8f wmma_bf(v16bf a, v16bf b, v8f c) {
  return __builtin_amdgcn_wmma_f32_16x16x32_bf16(false, a, false, b, (short)0, c, false, false);
}

// ---- TDM staging of one weight half (96 cols x 192 rows bf16) into LDS ----
// D# per ISA 8.3/8.4: 2D tensor, data_size=2B, tensor_dim0=192 (row length),
// tile_dim0=96 (cols), tile_dim1=192 (rows), tensor_dim0_stride=192.
static __device__ __forceinline__ void stage_weight_half(const __bf16* __restrict__ WT,
                                                         int half, __bf16* sm) {
#if __has_builtin(__builtin_amdgcn_tensor_load_to_lds)
  if ((threadIdx.x >> 5) == 0) {           // one wave issues the DMA
    const __bf16* gsrc = WT + half * kHalfK;
    unsigned long long ga = (unsigned long long)(uintptr_t)gsrc;
    unsigned int lds_addr = (unsigned int)(uintptr_t)sm;   // LDS aperture: low 32 bits
    u32x4 g0 = { 1u,                                   // count=1 (valid user D#)
                 lds_addr,
                 (unsigned int)ga,                     // global_addr[31:0]
                 (unsigned int)((ga >> 32) & 0x01FFFFFFu) | (2u << 30) };  // addr[56:32] | type=2
    i32x8 g1 = { 0x00010000,                           // data_size=1 -> 2 bytes
                 (int)(192u << 16),                    // tensor_dim0[15:0] at bits 63:48
                 (int)(192u << 16),                    // tensor_dim1[15:0] at bits 95:80
                 (int)((unsigned)kHalfK << 16),        // tile_dim0 at bits 127:112
                 192,                                  // tile_dim1 at bits 143:128
                 192,                                  // tensor_dim0_stride[31:0]
                 0, 0 };
    i32x4 gz4 = { 0, 0, 0, 0 };
    i32x8 gz8 = { 0, 0, 0, 0, 0, 0, 0, 0 };
    __builtin_amdgcn_tensor_load_to_lds(g0, g1, gz4, gz4, gz8, 0);  // 6-arg (clang-23 form)
    __builtin_amdgcn_s_wait_tensorcnt(0);
  }
#else
  // cooperative fallback: 16B per thread per iteration
  for (int i = threadIdx.x; i < (kC * kHalfK) / 8; i += blockDim.x) {
    int e = i * 8;
    int row = e / kHalfK, col = e - row * kHalfK;
    *reinterpret_cast<uint4*>(sm + row * kHalfK + col) =
        *reinterpret_cast<const uint4*>(WT + (size_t)row * kC + half * kHalfK + col);
  }
#endif
}

// ---------------- kernel 1: channel LayerNorm ----------------
__global__ void ln_kernel(const float* __restrict__ x,
                          const float* __restrict__ lnw,
                          const float* __restrict__ lnb,
                          float* __restrict__ xn,
                          __bf16* __restrict__ xnt) {
  int gid = blockIdx.x * blockDim.x + threadIdx.x;     // 0..kNTok-1, hw-major
  int b  = gid / kHW;
  int hw = gid - b * kHW;
  int h = hw / kW, w = hw - h * kW;
  size_t base = (size_t)b * kC * kHW + hw;
  float s = 0.f, s2 = 0.f;
#pragma unroll 8
  for (int c = 0; c < kC; ++c) {
    float v = x[base + (size_t)c * kHW];
    s += v; s2 += v * v;
  }
  float mu  = s * (1.f / kC);
  float var = s2 * (1.f / kC) - mu * mu;
  float ri  = rsqrtf(var + kEPS);
  size_t tbase = ((size_t)b * kT + (size_t)w * kH + h) * kC;   // W-major flatten
#pragma unroll 8
  for (int c = 0; c < kC; ++c) {
    float v = (x[base + (size_t)c * kHW] - mu) * ri * lnw[c] + lnb[c];
    xn[base + (size_t)c * kHW] = v;
    xnt[tbase + c] = f2bf(v);
  }
}

// ---------------- kernel 2: convert 4 weight matrices (192x192 f32) to bf16 ----------------
__global__ void cvt_wts_kernel(const float* __restrict__ w0, const float* __restrict__ w1,
                               const float* __restrict__ w2, const float* __restrict__ w3,
                               __bf16* __restrict__ dst) {
  int i = blockIdx.x * blockDim.x + threadIdx.x;       // 0..4*36864-1
  int sel = i / (kC * kC), j = i - sel * (kC * kC);
  const float* src = (sel == 0) ? w0 : (sel == 1) ? w1 : (sel == 2) ? w2 : w3;
  dst[i] = f2bf(src[j]);
}

// ---------------- kernel 3: pointwise GEMM  Out[m,o] = sum_c A[m,c] * W[o,c] + bias[o] ----------------
// M = kNTok, N = K = 192. One wave = 16 tokens x 192 outs (12 WMMA tiles).
// Weights staged to LDS in two halves via TDM; B frags come from LDS.
__global__ void gemm_pw_kernel(const __bf16* __restrict__ A,
                               const __bf16* __restrict__ WT,   // (o,c) row-major = B^T layout
                               const float* __restrict__ bias,
                               float* __restrict__ out) {
  __shared__ __bf16 smem[kC * kHalfK];                 // 36864 B
  int wave = blockIdx.x * (blockDim.x >> 5) + (threadIdx.x >> 5);
  int lane = threadIdx.x & 31;
  int m0 = wave * 16;
  v8f acc[12];
#pragma unroll
  for (int n = 0; n < 12; ++n) acc[n] = (v8f)(0.f);
#pragma unroll
  for (int hh = 0; hh < 2; ++hh) {
    __syncthreads();                                   // previous half fully consumed
    stage_weight_half(WT, hh, smem);
    __syncthreads();                                   // staged data visible
#pragma unroll
    for (int kkl = 0; kkl < 3; ++kkl) {
      int k0 = hh * kHalfK + kkl * 32;
      __builtin_prefetch(A + (size_t)(m0 + (lane & 15)) * kC + k0 + 32, 0, 1);
      v16bf a = load_a_frag(A, m0, kC, k0, lane);
#pragma unroll
      for (int n = 0; n < 12; ++n) {
        v16bf b = load_b_frag_lds(smem, n * 16, kkl * 32, lane);
        acc[n] = wmma_bf(a, b, acc[n]);
      }
    }
  }
  int mrow = m0 + ((lane & 16) ? 8 : 0);
#pragma unroll
  for (int n = 0; n < 12; ++n) {
    int o = n * 16 + (lane & 15);
    float bv = bias[o];
#pragma unroll
    for (int r = 0; r < 8; ++r)
      out[(size_t)(mrow + r) * kC + o] = acc[n][r] + bv;
  }
}

// ---------------- kernel 4: depthwise 3x3 conv (SAME, zero pad), bf16 out ----------------
__global__ void dwconv_kernel(const float* __restrict__ pw,
                              const float* __restrict__ wgt,   // (C,1,3,3)
                              const float* __restrict__ bias,
                              __bf16* __restrict__ out, int transposed) {
  int n = blockIdx.x;                 // token id over all batches
  int c = threadIdx.x;                // 0..191
  int b = n / kT;
  int t = n - b * kT;
  int w = t / kH, h = t - w * kH;     // t = w*H + h
  float acc = bias[c];
#pragma unroll
  for (int dh = -1; dh <= 1; ++dh) {
    int hh = h + dh;
    if (hh < 0 || hh >= kH) continue;
#pragma unroll
    for (int dw = -1; dw <= 1; ++dw) {
      int ww = w + dw;
      if (ww < 0 || ww >= kW) continue;
      int tn = ww * kH + hh;
      acc += wgt[c * 9 + (dh + 1) * 3 + (dw + 1)] *
             pw[((size_t)b * kT + tn) * kC + c];
    }
  }
  __bf16 v = f2bf(acc);
  if (transposed) out[((size_t)b * kC + c) * kT + t] = v;
  else            out[((size_t)b * kT + t) * kC + c] = v;
}

// ---------------- kernel 5: zero S (f32) and attT (bf16, padded 48x64) ----------------
__global__ void init_satt_kernel(float* __restrict__ S, __bf16* __restrict__ attT) {
  int i = blockIdx.x * blockDim.x + threadIdx.x;       // 0..98303
  if (i < kB * kHeads * kHS * kHS) S[i] = 0.f;
  attT[i] = __builtin_bit_cast(__bf16, (unsigned short)0);   // 32*48*64 = 98304
}

// ---------------- kernel 6: S[s,q] = sum_t K[t,s] Q[t,q]  (K-split + atomic f32) ----------------
__global__ void s_gemm_kernel(const __bf16* __restrict__ Kt,
                              const __bf16* __restrict__ Qt,
                              float* __restrict__ S) {
  int wave = blockIdx.x * (blockDim.x >> 5) + (threadIdx.x >> 5);
  int lane = threadIdx.x & 31;
  int bh    = wave / kKSplit;            // 0..31
  int chunk = wave - bh * kKSplit;
  int b = bh / kHeads, hd = bh - b * kHeads;
  const __bf16* kbase = Kt + (size_t)(b * kC + hd * kHS) * kT;
  const __bf16* qbase = Qt + (size_t)(b * kC + hd * kHS) * kT;
  int t0 = chunk * (kT / kKSplit);       // 256 tokens per chunk
  v8f acc[9];
#pragma unroll
  for (int i = 0; i < 9; ++i) acc[i] = (v8f)(0.f);
  for (int step = 0; step < (kT / kKSplit) / 32; ++step) {
    int k0 = t0 + step * 32;
    v16bf bfr[3];
#pragma unroll
    for (int j = 0; j < 3; ++j) bfr[j] = load_b_frag(qbase, j * 16, kT, k0, lane);
#pragma unroll
    for (int i = 0; i < 3; ++i) {
      v16bf a = load_a_frag(kbase, i * 16, kT, k0, lane);
#pragma unroll
      for (int j = 0; j < 3; ++j)
        acc[i * 3 + j] = wmma_bf(a, bfr[j], acc[i * 3 + j]);
    }
  }
  float* sdst = S + (size_t)bh * kHS * kHS;
#pragma unroll
  for (int i = 0; i < 3; ++i) {
#pragma unroll
    for (int j = 0; j < 3; ++j) {
      int col = j * 16 + (lane & 15);
      int rbase = i * 16 + ((lane & 16) ? 8 : 0);
#pragma unroll
      for (int r = 0; r < 8; ++r)
        unsafeAtomicAdd(&sdst[(rbase + r) * kHS + col], acc[i * 3 + j][r]);
    }
  }
}

// ---------------- kernel 7: softmax over q, write att^T (q-major, padded to 64) in bf16 ----------------
__global__ void softmax_kernel(const float* __restrict__ S,
                               const float* __restrict__ alpha,
                               __bf16* __restrict__ attT) {
  int row = blockIdx.x * blockDim.x + threadIdx.x;     // 0..32*48-1  (bh, s)
  if (row >= kB * kHeads * kHS) return;
  int bh = row / kHS, sidx = row - bh * kHS;
  const float* src = S + (size_t)row * kHS;
  float ia = 1.f / alpha[0];
  float mx = -3.4e38f;
  for (int q = 0; q < kHS; ++q) mx = fmaxf(mx, src[q]);
  float sum = 0.f;
  float e[kHS];
#pragma unroll 8
  for (int q = 0; q < kHS; ++q) { e[q] = __expf((src[q] - mx) * ia); sum += e[q]; }
  float inv = 1.f / sum;
  __bf16* dst = attT + (size_t)bh * kHS * 64;
#pragma unroll 8
  for (int q = 0; q < kHS; ++q) dst[q * 64 + sidx] = f2bf(e[q] * inv);
}

// ---------------- kernel 8: out[t,q] = sum_s V[t,s] att[s,q]   (M=tokens, N=K=48) ----------------
__global__ void av_gemm_kernel(const __bf16* __restrict__ V,       // token-major
                               const __bf16* __restrict__ attT,    // (bh,48,64) padded
                               __bf16* __restrict__ O) {           // token-major bf16
  int wave = blockIdx.x * (blockDim.x >> 5) + (threadIdx.x >> 5);
  int lane = threadIdx.x & 31;
  int ttile = wave >> 2;
  int hd = wave & 3;
  int m0 = ttile * 16;               // global token row
  int b = m0 >> 14;                  // /kT
  const __bf16* abase = V + hd * kHS;                          // row = token, lda = kC
  const __bf16* bt = attT + (size_t)(b * kHeads + hd) * kHS * 64;
  v8f acc[3];
#pragma unroll
  for (int j = 0; j < 3; ++j) acc[j] = (v8f)(0.f);
  // k chunk 0: s = 0..31 (full)
  {
    v16bf a = load_a_frag(abase, m0, kC, 0, lane);
#pragma unroll
    for (int j = 0; j < 3; ++j)
      acc[j] = wmma_bf(a, load_b_frag(bt, j * 16, 64, 0, lane), acc[j]);
  }
  // k chunk 1: s = 32..47 valid, 48..63 zero (attT zero-padded; A upper half zeroed)
  {
    const __bf16* p = abase + (size_t)(m0 + (lane & 15)) * kC + 32 + ((lane & 16) ? 8 : 0);
    FragU u;
    u.q[0] = *reinterpret_cast<const uint4*>(p);
    u.q[1] = make_uint4(0u, 0u, 0u, 0u);
#pragma unroll
    for (int j = 0; j < 3; ++j)
      acc[j] = wmma_bf(u.v, load_b_frag(bt, j * 16, 64, 32, lane), acc[j]);
  }
  int mrow = m0 + ((lane & 16) ? 8 : 0);
#pragma unroll
  for (int j = 0; j < 3; ++j) {
    int c = hd * kHS + j * 16 + (lane & 15);
#pragma unroll
    for (int r = 0; r < 8; ++r)
      O[(size_t)(mrow + r) * kC + c] = f2bf(acc[j][r]);
  }
}

// ---------------- kernel 9: final projection + bias + residual, write NCHW f32 ----------------
__global__ void final_gemm_kernel(const __bf16* __restrict__ A,    // attention out, token-major
                                  const __bf16* __restrict__ WT,   // f_w bf16 (o,c)
                                  const float* __restrict__ bias,  // f_b
                                  const float* __restrict__ xn,    // residual NCHW
                                  float* __restrict__ y) {
  __shared__ __bf16 smem[kC * kHalfK];
  int wave = blockIdx.x * (blockDim.x >> 5) + (threadIdx.x >> 5);
  int lane = threadIdx.x & 31;
  int m0 = wave * 16;
  v8f acc[12];
#pragma unroll
  for (int n = 0; n < 12; ++n) acc[n] = (v8f)(0.f);
#pragma unroll
  for (int hh = 0; hh < 2; ++hh) {
    __syncthreads();
    stage_weight_half(WT, hh, smem);
    __syncthreads();
#pragma unroll
    for (int kkl = 0; kkl < 3; ++kkl) {
      int k0 = hh * kHalfK + kkl * 32;
      __builtin_prefetch(A + (size_t)(m0 + (lane & 15)) * kC + k0 + 32, 0, 1);
      v16bf a = load_a_frag(A, m0, kC, k0, lane);
#pragma unroll
      for (int n = 0; n < 12; ++n) {
        v16bf b = load_b_frag_lds(smem, n * 16, kkl * 32, lane);
        acc[n] = wmma_bf(a, b, acc[n]);
      }
    }
  }
  int mrow = m0 + ((lane & 16) ? 8 : 0);
  int b = m0 >> 14;                         // batch
#pragma unroll
  for (int n = 0; n < 12; ++n) {
    int o = n * 16 + (lane & 15);
    float bv = bias[o];
    size_t obase = (size_t)(b * kC + o) * kHW;
#pragma unroll
    for (int r = 0; r < 8; ++r) {
      int t = (mrow + r) & (kT - 1);        // spatial linear index h*W+w
      y[obase + t] = acc[n][r] + bv + xn[obase + t];
    }
  }
}

// ---------------- host launch ----------------
extern "C" void kernel_launch(void* const* d_in, const int* in_sizes, int n_in,
                              void* d_out, int out_size, void* d_ws, size_t ws_size,
                              hipStream_t stream) {
  const float* x      = (const float*)d_in[0];
  const float* ln_w   = (const float*)d_in[1];
  const float* ln_b   = (const float*)d_in[2];
  const float* q_pw_w = (const float*)d_in[3];
  const float* q_pw_b = (const float*)d_in[4];
  const float* q_dw_w = (const float*)d_in[5];
  const float* q_dw_b = (const float*)d_in[6];
  const float* k_pw_w = (const float*)d_in[7];
  const float* k_pw_b = (const float*)d_in[8];
  const float* k_dw_w = (const float*)d_in[9];
  const float* k_dw_b = (const float*)d_in[10];
  const float* v_pw_w = (const float*)d_in[11];
  const float* v_pw_b = (const float*)d_in[12];
  const float* v_dw_w = (const float*)d_in[13];
  const float* v_dw_b = (const float*)d_in[14];
  const float* f_w    = (const float*)d_in[15];
  const float* f_b    = (const float*)d_in[16];
  const float* alpha  = (const float*)d_in[17];
  float* y = (float*)d_out;

  char* ws = (char*)d_ws;
  const size_t SZ_XN_F32 = (size_t)kNTok * kC * 4;     // 100663296
  const size_t SZ_TOK_BF = (size_t)kNTok * kC * 2;     //  50331648
  float*  xn   = (float*)(ws);                                 // NCHW f32
  __bf16* xnt  = (__bf16*)(ws + SZ_XN_F32);                    // token-major bf16
  float*  pw   = (float*)(ws + SZ_XN_F32 + SZ_TOK_BF);         // pw scratch f32
  __bf16* Qt   = (__bf16*)(ws + 2 * SZ_XN_F32 + SZ_TOK_BF);    // channel-major bf16
  __bf16* Kt   = (__bf16*)(ws + 2 * SZ_XN_F32 + 2 * SZ_TOK_BF);
  __bf16* Vb   = (__bf16*)(ws + 2 * SZ_XN_F32 + 3 * SZ_TOK_BF);
  char*   tail = ws + 2 * SZ_XN_F32 + 4 * SZ_TOK_BF;
  float*  Sb   = (float*)(tail);                               // 32*48*48 f32
  __bf16* attT = (__bf16*)(tail + 294912);                     // 32*48*64 bf16
  __bf16* wts  = (__bf16*)(tail + 294912 + 196608);            // 4 x 36864 bf16
  __bf16* wq = wts, *wk = wts + kC * kC, *wv = wts + 2 * kC * kC, *wf = wts + 3 * kC * kC;
  __bf16* Ob = (__bf16*)pw;                                    // reuse pw scratch

  ln_kernel<<<kNTok / 256, 256, 0, stream>>>(x, ln_w, ln_b, xn, xnt);
  cvt_wts_kernel<<<(4 * kC * kC) / 256, 256, 0, stream>>>(q_pw_w, k_pw_w, v_pw_w, f_w, wts);

  gemm_pw_kernel<<<(kNTok / 16) / 8, 256, 0, stream>>>(xnt, wq, q_pw_b, pw);
  dwconv_kernel<<<kNTok, kC, 0, stream>>>(pw, q_dw_w, q_dw_b, Qt, 1);
  gemm_pw_kernel<<<(kNTok / 16) / 8, 256, 0, stream>>>(xnt, wk, k_pw_b, pw);
  dwconv_kernel<<<kNTok, kC, 0, stream>>>(pw, k_dw_w, k_dw_b, Kt, 1);
  gemm_pw_kernel<<<(kNTok / 16) / 8, 256, 0, stream>>>(xnt, wv, v_pw_b, pw);
  dwconv_kernel<<<kNTok, kC, 0, stream>>>(pw, v_dw_w, v_dw_b, Vb, 0);

  init_satt_kernel<<<(kB * kHeads * kHS * 64) / 256, 256, 0, stream>>>(Sb, attT);
  s_gemm_kernel<<<(kB * kHeads * kKSplit) / 8, 256, 0, stream>>>(Kt, Qt, Sb);
  softmax_kernel<<<(kB * kHeads * kHS + 255) / 256, 256, 0, stream>>>(Sb, alpha, attT);
  av_gemm_kernel<<<((kNTok / 16) * kHeads) / 8, 256, 0, stream>>>(Vb, attT, Ob);
  final_gemm_kernel<<<(kNTok / 16) / 8, 256, 0, stream>>>(Ob, wf, f_b, xn, y);
}
